// EdgeQSim_68453188763808
// MI455X (gfx1250) — matrix operand order repaired and checked by reading, not compiled, not run on hardware
//
#include <hip/hip_runtime.h>
#include <hip/hip_bf16.h>
#include <stdint.h>

// ---------------------------------------------------------------------------
// EdgeQSim on MI455X (gfx1250): bf16 WMMA pipeline with async-LDS staging
//   B=32, N=512, OBJ=2048, QD=1024, M=B*N=16384, E gathered edges
// ---------------------------------------------------------------------------

typedef __attribute__((ext_vector_type(16))) __bf16 v16bf;
typedef __attribute__((ext_vector_type(8)))  __bf16 v8bf;
typedef __attribute__((ext_vector_type(8)))  float  v8f;

static constexpr int kB   = 32;
static constexpr int kN   = 512;
static constexpr int kM   = kB * kN;     // 16384 rows
static constexpr int kObj = 2048;
static constexpr int kQ   = 1024;

// workspace layout (bytes, all 256-aligned)
static constexpr size_t OFF_W1A = 0;                                   // 1024x2048 bf16
static constexpr size_t OFF_W2  = OFF_W1A + (size_t)1024 * 2048 * 2;   // 1024x1024 bf16
static constexpr size_t OFF_S1  = OFF_W2  + (size_t)1024 * 1024 * 2;   // 1024 f32
static constexpr size_t OFF_QC  = OFF_S1  + (size_t)1024 * 4;          // 32x1024 f32
static constexpr size_t OFF_H1  = OFF_QC  + (size_t)32 * 1024 * 4;     // 16384x1024 bf16
static constexpr size_t OFF_H2  = OFF_H1  + (size_t)kM * 1024 * 2;     // 16384x1024 bf16
static constexpr size_t OFF_X   = OFF_H2  + (size_t)kM * 1024 * 2;     // 16384x2048 bf16
static constexpr size_t OFF_S   = OFF_X;                               // overlay: 32x512x512 f32 (X is dead)

__device__ __forceinline__ unsigned short f2bf_bits(float x) {
    unsigned u = __builtin_bit_cast(unsigned, x);
    unsigned r = (u + 0x7FFFu + ((u >> 16) & 1u)) >> 16;   // round-to-nearest-even
    return (unsigned short)r;
}
__device__ __forceinline__ __bf16 f2bf(float x) {
    unsigned short h = f2bf_bits(x);
    return __builtin_bit_cast(__bf16, h);
}

// async copy of 16 bytes global -> LDS (ASYNCcnt-tracked, no VGPR transit)
__device__ __forceinline__ void async_g2l_b128(unsigned ldsOff, const void* g) {
    asm volatile("global_load_async_to_lds_b128 %0, %1, off"
                 :: "v"(ldsOff), "v"((unsigned long long)(size_t)g)
                 : "memory");
}
__device__ __forceinline__ void wait_asynccnt0() {
    asm volatile("s_wait_asynccnt 0x0" ::: "memory");
}

// ---------------------------------------------------------------------------
// Weight-norm prep: W[row,:Kout] = bf16(v[row,:Kout] * g[row]/||v[row,:K]||)
// Also emits s[row] = g[row]/||v[row]|| when sOut != nullptr.
// ---------------------------------------------------------------------------
__global__ __launch_bounds__(256)
void wnorm_kernel(const float* __restrict__ v, const float* __restrict__ g,
                  __bf16* __restrict__ W, float* __restrict__ sOut,
                  int K, int Kout) {
    __shared__ float red[256];
    __shared__ float sSh;
    const int row = blockIdx.x;
    const int tid = threadIdx.x;
    const float* vr = v + (size_t)row * K;
    float ss = 0.f;
    for (int i = tid; i < K; i += 256) { float x = vr[i]; ss += x * x; }
    red[tid] = ss;
    __syncthreads();
    for (int s = 128; s > 0; s >>= 1) {
        if (tid < s) red[tid] += red[tid + s];
        __syncthreads();
    }
    if (tid == 0) {
        float sc = g[row] / sqrtf(red[0]);
        sSh = sc;
        if (sOut) sOut[row] = sc;
    }
    __syncthreads();
    const float sc = sSh;
    __bf16* wr = W + (size_t)row * Kout;
    for (int i = tid; i < Kout; i += 256) wr[i] = f2bf(vr[i] * sc);
}

// ---------------------------------------------------------------------------
// qc[b,d] = b1[d] + s1[d] * dot(q[b,:], v1[d, 2048:3072])
// ---------------------------------------------------------------------------
__global__ __launch_bounds__(256)
void qc_kernel(const float* __restrict__ q, const float* __restrict__ v1,
               const float* __restrict__ s1, const float* __restrict__ b1,
               float* __restrict__ qc) {
    const int tid = threadIdx.x;
    const int d = blockIdx.x * 8 + (tid >> 5);
    const int b = tid & 31;
    const float* vr = v1 + (size_t)d * 3072 + 2048;
    const float* qr = q + (size_t)b * kQ;
    float acc = 0.f;
#pragma unroll 4
    for (int c = 0; c < kQ; ++c) acc += qr[c] * vr[c];
    qc[(size_t)b * 1024 + d] = b1[d] + s1[d] * acc;
}

// ---------------------------------------------------------------------------
// fp32 -> bf16 conversion, 4 elems/thread (exact grid, no tail)
// ---------------------------------------------------------------------------
__global__ __launch_bounds__(256)
void cvt_bf16_kernel(const float4* __restrict__ in, uint2* __restrict__ out) {
    size_t i = (size_t)blockIdx.x * 256 + threadIdx.x;
    float4 f = in[i];
    uint2 p;
    p.x = (unsigned)f2bf_bits(f.x) | ((unsigned)f2bf_bits(f.y) << 16);
    p.y = (unsigned)f2bf_bits(f.z) | ((unsigned)f2bf_bits(f.w) << 16);
    out[i] = p;
}

// ---------------------------------------------------------------------------
// WMMA GEMM:  C[m,n] = sum_k A[m,k] * B[n,k]   (both row-major, ld == K)
//   MODE 0: +qc[(m/512)*1024 + n], relu, bf16 out (ldOut=1024)
//   MODE 1: +bias[n],              relu, bf16 out (ldOut=1024)
//   MODE 2: batched over blockIdx.z (512-row batches), f32 out (ldOut=512)
//
// Workgroup = 256 threads = 8 waves (2 along M x 4 along N), WG tile 128x128,
// wave tile 64x32.  K staged in 64-wide double-buffered LDS tiles filled with
// global_load_async_to_lds_b128 (ASYNCcnt), consumed via ds_load fragments.
// All dims are multiples of the tiles: no predication, EXEC stays all-1s.
// ---------------------------------------------------------------------------
template <int MODE>
__global__ __launch_bounds__(256)
void gemm_wmma_kernel(const __bf16* __restrict__ A, const __bf16* __restrict__ B,
                      const float* __restrict__ bias, void* __restrict__ Out,
                      int K, int ldOut) {
    __shared__ __align__(32) __bf16 As[2][128 * 64];   // 2 x 16 KB
    __shared__ __align__(32) __bf16 Bs[2][128 * 64];   // 2 x 16 KB

    const int tid  = threadIdx.x;
    const int lane = tid & 31;
    const int wv   = tid >> 5;            // 0..7
    const int wm   = wv & 1;              // 2 waves along M
    const int wn   = wv >> 1;             // 4 waves along N
    const int rowWG = blockIdx.y * 128;   // WG tile origin (global)
    const int colWG = blockIdx.x * 128;
    const int waveRow = wm * 64;          // within WG tile
    const int waveCol = wn * 32;

    const __bf16* Ab = A;
    const __bf16* Bb = B;
    float*  Of = (float*)Out;
    __bf16* Oh = (__bf16*)Out;
    if (MODE == 2) {
        size_t bo = (size_t)blockIdx.z * 512 * (size_t)K;
        Ab += bo;
        Bb += bo;
        Of += (size_t)blockIdx.z * 512 * 512;
    }

    const v8f vzero = {0.f, 0.f, 0.f, 0.f, 0.f, 0.f, 0.f, 0.f};
    v8f acc[4][2];
#pragma unroll
    for (int m = 0; m < 4; ++m)
#pragma unroll
        for (int n = 0; n < 2; ++n) acc[m][n] = vzero;

    const int lr   = lane & 15;           // row/col within 16x16 tile
    const int ks8  = (lane >> 4) * 8;     // A fragment K sub-offset
    const int ks16 = (lane >> 4) * 16;    // unused; B fragments also use ks8

    // stage one 128x64 A tile + 128x64 B tile into LDS buffer `buf`
    auto stage = [&](int buf, int kt) {
#pragma unroll
        for (int p = 0; p < 4; ++p) {
            const int s   = tid + p * 256;    // 0..1023 16B-segments
            const int row = s >> 3;           // 0..127
            const int el  = (s & 7) * 8;      // element offset within 64-wide row
            async_g2l_b128((unsigned)(size_t)&As[buf][row * 64 + el],
                           Ab + (size_t)(rowWG + row) * K + kt + el);
            async_g2l_b128((unsigned)(size_t)&Bs[buf][row * 64 + el],
                           Bb + (size_t)(colWG + row) * K + kt + el);
        }
    };

    const int nStages = K >> 6;           // K / 64
    stage(0, 0);
    int buf = 0;
    for (int st = 0; st < nStages; ++st) {
        wait_asynccnt0();                 // my async writes for `buf` done
        __syncthreads();                  // everyone's done; prev buf free
        if (st + 1 < nStages) stage(buf ^ 1, (st + 1) * 64);

#pragma unroll
        for (int c = 0; c < 2; ++c) {     // two 32-wide K chunks per stage
            v16bf af[4];
#pragma unroll
            for (int m = 0; m < 4; ++m) {
                const __bf16* ap =
                    &As[buf][(waveRow + m * 16 + lr) * 64 + c * 32 + ks8];
                v8bf lo = *(const v8bf*)ap;          // k .. k+7
                v8bf hi = *(const v8bf*)(ap + 16);   // k+16 .. k+23
#pragma unroll
                for (int i = 0; i < 8; ++i) { af[m][i] = lo[i]; af[m][i + 8] = hi[i]; }
            }
            v16bf bfr[2];
#pragma unroll
            for (int n = 0; n < 2; ++n) {
                const __bf16* bp =
                    &Bs[buf][(waveCol + n * 16 + lr) * 64 + c * 32 + ks8];
                v8bf lo = *(const v8bf*)bp;
                v8bf hi = *(const v8bf*)(bp + 16);
#pragma unroll
                for (int i = 0; i < 8; ++i) { bfr[n][i] = lo[i]; bfr[n][i + 8] = hi[i]; }
            }
#pragma unroll
            for (int m = 0; m < 4; ++m)
#pragma unroll
                for (int n = 0; n < 2; ++n)
                    acc[m][n] = __builtin_amdgcn_wmma_f32_16x16x32_bf16(
                        false, af[m], false, bfr[n], (short)0, acc[m][n],
                        false, false);
        }
        buf ^= 1;
    }

    // epilogue: C/D layout -> row = v + (lane/16)*8, col = lane%16
    const int rHalf = (lane >> 4) * 8;
#pragma unroll
    for (int m = 0; m < 4; ++m) {
#pragma unroll
        for (int n = 0; n < 2; ++n) {
#pragma unroll
            for (int v = 0; v < 8; ++v) {
                const int r = rowWG + waveRow + m * 16 + rHalf + v;
                const int c = colWG + waveCol + n * 16 + lr;
                float val = acc[m][n][v];
                if (MODE == 0) {
                    val += bias[(size_t)(r >> 9) * 1024 + c];  // qc[batch, c]
                    val = val > 0.f ? val : 0.f;
                    Oh[(size_t)r * ldOut + c] = f2bf(val);
                } else if (MODE == 1) {
                    val += bias[c];
                    val = val > 0.f ? val : 0.f;
                    Oh[(size_t)r * ldOut + c] = f2bf(val);
                } else {
                    Of[(size_t)r * ldOut + c] = val;
                }
            }
        }
    }
}

// ---------------------------------------------------------------------------
// gather: out[e] = S[indexes[e]]   (S is 32*512*512, L2-resident)
// ---------------------------------------------------------------------------
__global__ __launch_bounds__(256)
void gather_kernel(const int* __restrict__ idx, const float* __restrict__ S,
                   float* __restrict__ out, int E) {
    int e = blockIdx.x * 256 + threadIdx.x;
    if (e < E) out[e] = S[idx[e]];
}

// ---------------------------------------------------------------------------
extern "C" void kernel_launch(void* const* d_in, const int* in_sizes, int n_in,
                              void* d_out, int out_size, void* d_ws, size_t ws_size,
                              hipStream_t stream) {
    const float* node = (const float*)d_in[0];   // [32,512,2048]
    const float* qf   = (const float*)d_in[1];   // [32,1024]
    const int*   idx  = (const int*)d_in[2];     // [E]
    const float* v1   = (const float*)d_in[3];   // [1024,3072]
    const float* g1   = (const float*)d_in[4];   // [1024]
    const float* b1   = (const float*)d_in[5];   // [1024]
    const float* v2   = (const float*)d_in[6];   // [1024,1024]
    const float* g2   = (const float*)d_in[7];   // [1024]
    const float* b2   = (const float*)d_in[8];   // [1024]
    const int E = in_sizes[2];

    char* ws = (char*)d_ws;
    __bf16* W1a = (__bf16*)(ws + OFF_W1A);
    __bf16* W2  = (__bf16*)(ws + OFF_W2);
    float*  s1  = (float*)(ws + OFF_S1);
    float*  qc  = (float*)(ws + OFF_QC);
    __bf16* h1  = (__bf16*)(ws + OFF_H1);
    __bf16* h2  = (__bf16*)(ws + OFF_H2);
    __bf16* X   = (__bf16*)(ws + OFF_X);
    float*  S   = (float*)(ws + OFF_S);

    // 1) weight-norm -> bf16 weights (+ s1 scales for the q-part fold)
    wnorm_kernel<<<1024, 256, 0, stream>>>(v1, g1, W1a, s1, 3072, 2048);
    wnorm_kernel<<<1024, 256, 0, stream>>>(v2, g2, W2, nullptr, 1024, 1024);

    // 2) fold the broadcast q-half of GEMM1 into a per-(batch,channel) bias
    qc_kernel<<<128, 256, 0, stream>>>(qf, v1, s1, b1, qc);

    // 3) node feats fp32 -> bf16
    cvt_bf16_kernel<<<(kM * kObj) / (256 * 4), 256, 0, stream>>>(
        (const float4*)node, (uint2*)X);

    // 4) h1 = relu(X @ W1aT + qc)        M=16384 N=1024 K=2048
    gemm_wmma_kernel<0><<<dim3(1024 / 128, kM / 128, 1), 256, 0, stream>>>(
        X, W1a, qc, (void*)h1, 2048, 1024);

    // 5) h2 = relu(h1 @ W2T + b2)        M=16384 N=1024 K=1024
    gemm_wmma_kernel<1><<<dim3(1024 / 128, kM / 128, 1), 256, 0, stream>>>(
        h1, W2, b2, (void*)h2, 1024, 1024);

    // 6) S_b = h2_b @ h2_bT per batch    M=N=512 K=1024, 32 batches
    gemm_wmma_kernel<2><<<dim3(512 / 128, 512 / 128, kB), 256, 0, stream>>>(
        h2, h2, nullptr, (void*)S, 1024, 512);

    // 7) gather requested edges
    gather_kernel<<<(E + 255) / 256, 256, 0, stream>>>(idx, S, (float*)d_out, E);
}
